// graph_undirected_sep_63651415327267
// MI455X (gfx1250) — matrix operand order repaired and checked by reading, not compiled, not run on hardware
//
#include <hip/hip_runtime.h>
#include <hip/hip_bf16.h>

typedef __attribute__((ext_vector_type(16))) _Float16 v16h;
typedef __attribute__((ext_vector_type(8)))  _Float16 v8h;
typedef __attribute__((ext_vector_type(8)))  float    v8f;

#define N_MOD 3
#define M_SUB 2048
#define DIM   64
#define KBLK  (N_MOD * N_MOD)        // 9
#define NTOT  (N_MOD * M_SUB)        // 6144
#define ALPHA 3.0f
#define TOPK  20

// Branch-free relu(tanh(x)): for y = max(x,0), tanh(y) = 1 - 2/(exp(2y)+1).
// exp(2y) = exp2(2*log2(e)*y) -> single v_exp_f32. Overflow -> +inf -> rcp -> 0
// -> result 1.0, so no special-casing needed. Monotone in x, so downstream
// top-k ordering matches exact tanh.
__device__ __forceinline__ float relu_tanh(float x) {
    float y = fmaxf(x, 0.0f);
    float e = __builtin_amdgcn_exp2f(y * 2.885390082f /* 2*log2(e) */);
    float r = __builtin_amdgcn_rcpf(e + 1.0f);
    return fmaf(-2.0f, r, 1.0f);
}

// Branch-free full-range tanh via the same identity + copysign.
__device__ __forceinline__ float fast_tanh(float x) {
    float a = fabsf(x);
    float e = __builtin_amdgcn_exp2f(a * 2.885390082f);
    float r = __builtin_amdgcn_rcpf(e + 1.0f);
    float t = fmaf(-2.0f, r, 1.0f);
    return copysignf(t, x);
}

// ---------------------------------------------------------------------------
// Kernel 1: h1[k] = tanh(a*(emb[k] @ W[k]^T + b[k]))
//           h2[k] = tanh(a*(emb[perm(k)] @ W[k]^T + b[k])), perm(i*n+j)=j*n+i
// One 64-thread block per (k, m) row; emb rows staged in LDS; f16 output.
// ---------------------------------------------------------------------------
__global__ void __launch_bounds__(DIM)
compute_h_kernel(const float* __restrict__ emb,
                 const float* __restrict__ W,
                 const float* __restrict__ b,
                 _Float16* __restrict__ h1,
                 _Float16* __restrict__ h2) {
    __shared__ float e1[DIM];
    __shared__ float e2[DIM];
    int bid = blockIdx.x;                 // k*M_SUB + m
    int k   = bid >> 11;
    int m   = bid & (M_SUB - 1);
    int i   = k / N_MOD, j = k % N_MOD;
    int ksrc = j * N_MOD + i;             // permuted block for h2
    int tid = threadIdx.x;

    e1[tid] = emb[((size_t)k    * M_SUB + m) * DIM + tid];
    e2[tid] = emb[((size_t)ksrc * M_SUB + m) * DIM + tid];
    __syncthreads();

    const float* wrow = W + ((size_t)k * DIM + tid) * DIM;  // W[k][e][:]
    float bias = b[k * DIM + tid];
    float a1 = bias, a2 = bias;
#pragma unroll
    for (int d = 0; d < DIM; ++d) {
        float w = wrow[d];
        a1 = fmaf(e1[d], w, a1);
        a2 = fmaf(e2[d], w, a2);
    }
    size_t o = ((size_t)k * M_SUB + m) * DIM + tid;
    h1[o] = (_Float16)fast_tanh(ALPHA * a1);
    h2[o] = (_Float16)fast_tanh(ALPHA * a2);
}

// ---------------------------------------------------------------------------
// Kernel 2: per block k, A = relu(tanh(a * h1 @ h2^T)) via v_wmma_f32_16x16x32_f16.
// One wave computes a 16x64 strip (4 N-tiles, 2 K-steps each = 8 WMMAs),
// storing straight into the assembled (6144 x 6144) adjacency.
// ---------------------------------------------------------------------------
__global__ void __launch_bounds__(256)
adj_wmma_kernel(const _Float16* __restrict__ h1,
                const _Float16* __restrict__ h2,
                float* __restrict__ out) {
    int wave = (blockIdx.x << 3) + (threadIdx.x >> 5);   // 8 waves / block
    int lane = threadIdx.x & 31;
    int k      = wave >> 12;          // / (128 mtiles * 32 nstrips)
    int rem    = wave & 4095;
    int mtile  = rem >> 5;
    int nstrip = rem & 31;
    int i = k / N_MOD, j = k % N_MOD;

    int mrow  = (mtile << 4) + (lane & 15);
    int khalf = lane >> 4;            // lane-group K split

    // A fragment, contiguous k order used here (B matches): lanes 0-15 hold row M=lane, K=0..15 contiguous;
    // lanes 16-31 hold K=16..31. Two K-steps cover K=0..63.
    const _Float16* arow = h1 + ((size_t)k * M_SUB + mrow) * DIM;
    v16h afrag[2];
#pragma unroll
    for (int ks = 0; ks < 2; ++ks) {
        int e0 = ks * 32 + khalf * 16;
        v8h p = *(const v8h*)(arow + e0);
        v8h q = *(const v8h*)(arow + e0 + 8);
#pragma unroll
        for (int t = 0; t < 8; ++t) { afrag[ks][t] = p[t]; afrag[ks][t + 8] = q[t]; }
    }

    int nbase = nstrip << 6;
#pragma unroll
    for (int nt = 0; nt < 4; ++nt) {
        int ncol = nbase + (nt << 4) + (lane & 15);
        // B column N = h2 row ncol; lanes 0-15: K=0..15 contiguous, lanes 16-31: K=16..31.
        const _Float16* brow = h2 + ((size_t)k * M_SUB + ncol) * DIM;
        v8f acc = {};
#pragma unroll
        for (int ks = 0; ks < 2; ++ks) {
            v16h bfrag = *(const v16h*)(brow + ks * 32 + khalf * 16);
            acc = __builtin_amdgcn_wmma_f32_16x16x32_f16(
                false, afrag[ks], false, bfrag, (short)0, acc, false, false);
        }
        // C/D layout: VGPR r, lanes 0-15 -> M=r, lanes 16-31 -> M=r+8; N = lane&15.
        int gc = j * M_SUB + ncol;
#pragma unroll
        for (int r = 0; r < 8; ++r) {
            int mm = (mtile << 4) + r + (khalf << 3);
            out[(size_t)(i * M_SUB + mm) * NTOT + gc] = relu_tanh(ALPHA * acc[r]);
        }
    }
}

// ---------------------------------------------------------------------------
// Kernel 3: per-row top-20 mask, in place. Values are >= 0 so float bits are
// order-monotonic: 4-round byte radix-select via LDS histograms gives the
// exact 20th-largest bit pattern + tie budget; ties kept in ascending index
// order (matches top_k scatter). One 256-thread block per row, 24 elems/thread.
// ---------------------------------------------------------------------------
__global__ void __launch_bounds__(256)
topk_mask_kernel(float* __restrict__ adj) {
    __shared__ unsigned hist[256];
    __shared__ unsigned scanbuf[256];
    __shared__ unsigned sPrefix, sKrem;
    const int CH = NTOT / 256;           // 24 contiguous elements per thread
    int tid  = threadIdx.x;
    float* rp = adj + (size_t)blockIdx.x * NTOT;
    int base = tid * CH;

    if (tid == 0) { sPrefix = 0u; sKrem = TOPK; }
    __syncthreads();

    for (int round = 0; round < 4; ++round) {
        hist[tid] = 0u;
        __syncthreads();
        int shift = 24 - 8 * round;
        unsigned pfx = sPrefix;
        for (int c = 0; c < CH; ++c) {
            unsigned u = __float_as_uint(rp[base + c]);
            bool match = (round == 0) || ((u >> (shift + 8)) == pfx);
            if (match) atomicAdd(&hist[(u >> shift) & 0xFFu], 1u);
        }
        __syncthreads();
        if (tid == 0) {
            unsigned cum = 0u, krem = sKrem, p = sPrefix;
            for (int bx = 255; bx >= 0; --bx) {
                unsigned c = hist[bx];
                if (cum + c >= krem) {
                    sPrefix = (p << 8) | (unsigned)bx;
                    sKrem   = krem - cum;
                    break;
                }
                cum += c;
            }
        }
        __syncthreads();
    }

    unsigned T   = sPrefix;   // bit pattern of the 20th-largest value
    unsigned req = sKrem;     // how many elements equal to T to keep (>=1)

    unsigned eq = 0u;
    for (int c = 0; c < CH; ++c)
        if (__float_as_uint(rp[base + c]) == T) eq++;
    scanbuf[tid] = eq;
    __syncthreads();
    if (tid == 0) {                       // exclusive scan over 256 chunk counts
        unsigned run = 0u;
        for (int t = 0; t < 256; ++t) { unsigned v = scanbuf[t]; scanbuf[t] = run; run += v; }
    }
    __syncthreads();
    unsigned ebase = scanbuf[tid];
    unsigned le = 0u;
    for (int c = 0; c < CH; ++c) {
        float v = rp[base + c];
        unsigned u = __float_as_uint(v);
        bool keep = (u > T);
        if (u == T) { keep = (ebase + le) < req; le++; }
        rp[base + c] = keep ? v : 0.0f;
    }
}

// ---------------------------------------------------------------------------
extern "C" void kernel_launch(void* const* d_in, const int* in_sizes, int n_in,
                              void* d_out, int out_size, void* d_ws, size_t ws_size,
                              hipStream_t stream) {
    // inputs: 0=idx(int32, unused beyond shape), 1=emb, 2=W, 3=b  (all fp32)
    const float* emb = (const float*)d_in[1];
    const float* W   = (const float*)d_in[2];
    const float* b   = (const float*)d_in[3];
    float* out = (float*)d_out;

    _Float16* h1 = (_Float16*)d_ws;                       // 9*2048*64 halves
    _Float16* h2 = h1 + (size_t)KBLK * M_SUB * DIM;       // 9*2048*64 halves (~4.5MB total)

    // 1) hidden features in f16
    compute_h_kernel<<<KBLK * M_SUB, DIM, 0, stream>>>(emb, W, b, h1, h2);

    // 2) WMMA block GEMM + relu(tanh) -> assembled adjacency
    //    waves = 9 blocks * 128 mtiles * 32 nstrips = 36864; 8 waves/block
    adj_wmma_kernel<<<36864 / 8, 256, 0, stream>>>(h1, h2, out);

    // 3) per-row top-20 mask, in place
    topk_mask_kernel<<<NTOT, 256, 0, stream>>>(out);
}
